// GNNModel_32358283608543
// MI455X (gfx1250) — matrix operand order repaired and checked by reading, not compiled, not run on hardware
//
#include <hip/hip_runtime.h>
#include <hip/hip_bf16.h>

typedef __attribute__((ext_vector_type(16))) __bf16 v16bf;
typedef __attribute__((ext_vector_type(8)))  float  v8f;

// ---------------------------------------------------------------------------
// Relaxed, device-scope float atomic add -> global_atomic_add_f32 on gfx1250
// ---------------------------------------------------------------------------
__device__ __forceinline__ void fatomic_add(float* p, float v) {
    __hip_atomic_fetch_add(p, v, __ATOMIC_RELAXED, __HIP_MEMORY_SCOPE_AGENT);
}

// ---------------------------------------------------------------------------
// Zero fill
// ---------------------------------------------------------------------------
__global__ void zero_f32(float* __restrict__ p, long long n) {
    long long i = (long long)blockIdx.x * blockDim.x + threadIdx.x;
    long long stride = (long long)gridDim.x * blockDim.x;
    for (; i < n; i += stride) p[i] = 0.0f;
}

// ---------------------------------------------------------------------------
// One-time f32 -> bf16 weight conversion (weights stay L2/LDS resident after)
// ---------------------------------------------------------------------------
__global__ void f32_to_bf16(const float* __restrict__ s, __bf16* __restrict__ d, int n) {
    int i = blockIdx.x * blockDim.x + threadIdx.x;
    if (i < n) d[i] = (__bf16)s[i];
}

// ---------------------------------------------------------------------------
// Edge scatter: one wave (32 lanes) per edge; lane j moves float4 #j of the
// 128-float feature row. agg[dst] += x[src]; cnt[dst] += 1 (lane 0 only).
// Feature matrices (51.2 MB) are L2-resident; this is L2-atomic bound.
// ---------------------------------------------------------------------------
__global__ __launch_bounds__(256) void edge_scatter(
    const float* __restrict__ x, const int* __restrict__ src,
    const int* __restrict__ dst, float* __restrict__ agg,
    float* __restrict__ cnt, long long E, int withCount)
{
    long long t = (long long)blockIdx.x * blockDim.x + threadIdx.x;
    if (t >= E * 32) return;
    long long e = t >> 5;
    int j = (int)(t & 31);
    int s = src[e], d = dst[e];
    const float4 v = ((const float4*)(x + (size_t)s * 128))[j];
    float* a = agg + (size_t)d * 128 + j * 4;
    fatomic_add(a + 0, v.x);
    fatomic_add(a + 1, v.y);
    fatomic_add(a + 2, v.z);
    fatomic_add(a + 3, v.w);
    if (withCount && j == 0) fatomic_add(cnt + d, 1.0f);
}

// ---------------------------------------------------------------------------
// A-fragment packer (f32 memory -> bf16 WMMA fragment halves).
// A 16x32 bf16 (lane = M row): elements e=0..15 <-> k = chunk*32 + hf*16 + e
// ---------------------------------------------------------------------------
__device__ __forceinline__ void pack8(v16bf& d, int base, float4 a, float4 b, float s) {
    d[base + 0] = (__bf16)(a.x * s); d[base + 1] = (__bf16)(a.y * s);
    d[base + 2] = (__bf16)(a.z * s); d[base + 3] = (__bf16)(a.w * s);
    d[base + 4] = (__bf16)(b.x * s); d[base + 5] = (__bf16)(b.y * s);
    d[base + 6] = (__bf16)(b.z * s); d[base + 7] = (__bf16)(b.w * s);
}

// ---------------------------------------------------------------------------
// Fused SAGEConv GEMM:
//   out[m, :] = relu( (agg[m,:]/max(cnt[m],1)) @ Wl^T + bl + xroot[m,:] @ Wr^T )
// 8 waves/block; both bf16 weight matrices staged once per block in LDS
// (64 KB of the 320 KB/WGP). One wave per 16-row M tile; K = 128 (4 chunks
// of 32); N = 128 (8 tiles). A fragments in registers across all N tiles.
// B 32x16 bf16 (lane = N col): elements e=0..15 <-> k = chunk*32 + hf*16 + e
// -> 16 contiguous bf16 in LDS = one aligned v16bf (two ds_load_b128).
// Epilogue: wave-uniform full-tile test -> unguarded store clause (no EXEC
// juggling); guarded path only for a genuine tail tile.
// ---------------------------------------------------------------------------
__global__ __launch_bounds__(256) void sage_gemm(
    const float* __restrict__ agg, const float* __restrict__ cnt,
    const float* __restrict__ xroot,
    const __bf16* __restrict__ Wl, const float* __restrict__ bl,
    const __bf16* __restrict__ Wr,
    float* __restrict__ out, int nTiles, int nRows)
{
    __shared__ __bf16 sWl[128 * 128];   // 32 KB
    __shared__ __bf16 sWr[128 * 128];   // 32 KB

    // Cooperative weight stage (all 256 threads, before any wave exits)
    {
        const uint4* gl = (const uint4*)Wl;   // 2048 x 16B per matrix
        const uint4* gr = (const uint4*)Wr;
        uint4* ll = (uint4*)sWl;
        uint4* lr = (uint4*)sWr;
        for (int i = threadIdx.x; i < 2048; i += 256) {
            ll[i] = gl[i];
            lr[i] = gr[i];
        }
    }
    __syncthreads();

    int wave = threadIdx.x >> 5;
    int tile_m = blockIdx.x * 8 + wave;
    if (tile_m >= nTiles) return;                    // wave-uniform exit

    int lane = threadIdx.x & 31;
    int r  = lane & 15;
    int hf = lane >> 4;

    int row = tile_m * 16 + r;
    if (row >= nRows) row = nRows - 1;               // clamp (loads only)
    float invc = 1.0f / fmaxf(cnt[row], 1.0f);

    const float* arow = agg   + (size_t)row * 128;
    const float* xrow = xroot + (size_t)row * 128;

    v16bf Am[4], Ax[4];
#pragma unroll
    for (int c = 0; c < 4; ++c) {
        int base = c * 32 + hf * 16;
        float4 p0 = *(const float4*)(arow + base);
        float4 p1 = *(const float4*)(arow + base + 4);
        float4 p2 = *(const float4*)(arow + base + 8);
        float4 p3 = *(const float4*)(arow + base + 12);
        pack8(Am[c], 0, p0, p1, invc);
        pack8(Am[c], 8, p2, p3, invc);
        float4 q0 = *(const float4*)(xrow + base);
        float4 q1 = *(const float4*)(xrow + base + 4);
        float4 q2 = *(const float4*)(xrow + base + 8);
        float4 q3 = *(const float4*)(xrow + base + 12);
        pack8(Ax[c], 0, q0, q1, 1.0f);
        pack8(Ax[c], 8, q2, q3, 1.0f);
    }

    // Preload per-lane biases for all 8 N tiles (one load clause)
    float bo[8];
#pragma unroll
    for (int nt = 0; nt < 8; ++nt) bo[nt] = bl[nt * 16 + r];

    const bool fullTile = (tile_m * 16 + 16) <= nRows;   // wave-uniform

    for (int nt = 0; nt < 8; ++nt) {
        int o = nt * 16 + r;                         // output column for this lane
        const __bf16* wl = sWl + (size_t)o * 128 + hf * 16;
        const __bf16* wr = sWr + (size_t)o * 128 + hf * 16;
        v8f acc = {};
#pragma unroll
        for (int c = 0; c < 4; ++c) {
            v16bf Bl = *(const v16bf*)(wl + c * 32);
            v16bf Br = *(const v16bf*)(wr + c * 32);
            acc = __builtin_amdgcn_wmma_f32_16x16x32_bf16(
                false, Am[c], false, Bl, (short)0, acc, false, false);
            acc = __builtin_amdgcn_wmma_f32_16x16x32_bf16(
                false, Ax[c], false, Br, (short)0, acc, false, false);
        }
        float* orow = out + (size_t)(tile_m * 16 + hf * 8) * 128 + o;
        float b = bo[nt];
        if (fullTile) {
            // D layout: VGPR k -> M row (tile_m*16 + hf*8 + k), col o
#pragma unroll
            for (int k = 0; k < 8; ++k)
                orow[(size_t)k * 128] = fmaxf(acc[k] + b, 0.0f);
        } else {
#pragma unroll
            for (int k = 0; k < 8; ++k) {
                int m = tile_m * 16 + hf * 8 + k;
                if (m < nRows)
                    out[(size_t)m * 128 + o] = fmaxf(acc[k] + b, 0.0f);
            }
        }
    }
}

// ---------------------------------------------------------------------------
// Classifier head: per-graph offsets from batch_vector, gather targets,
// 128 -> 64 -> 1 MLP. One block; B <= 128.
// ---------------------------------------------------------------------------
__global__ __launch_bounds__(128) void classifier(
    const float* __restrict__ h, const int* __restrict__ batch,
    const int* __restrict__ tgt,
    const float* __restrict__ Wc1, const float* __restrict__ bc1,
    const float* __restrict__ Wc2, const float* __restrict__ bc2,
    float* __restrict__ out, int N, int B)
{
    __shared__ int cnts[128];
    __shared__ int offs[128];
    int t = threadIdx.x;
    cnts[t] = 0;
    __syncthreads();
    for (int i = t; i < N; i += blockDim.x)
        atomicAdd(&cnts[batch[i]], 1);
    __syncthreads();
    if (t == 0) {
        int s = 0;
        for (int g = 0; g < B; ++g) { offs[g] = s; s += cnts[g]; }
    }
    __syncthreads();
    if (t < B) {
        const float* row = h + (size_t)(tgt[t] + offs[t]) * 128;
        float acc2 = bc2[0];
        for (int j = 0; j < 64; ++j) {
            float s = bc1[j];
            const float* w = Wc1 + (size_t)j * 128;
            for (int k = 0; k < 128; ++k) s += w[k] * row[k];
            acc2 += Wc2[j] * fmaxf(s, 0.0f);
        }
        out[t] = acc2;
    }
}

// ---------------------------------------------------------------------------
extern "C" void kernel_launch(void* const* d_in, const int* in_sizes, int n_in,
                              void* d_out, int out_size, void* d_ws, size_t ws_size,
                              hipStream_t stream) {
    const float* x    = (const float*)d_in[0];
    const int*   eidx = (const int*)  d_in[1];
    const int*   tgt  = (const int*)  d_in[2];
    const int*   bat  = (const int*)  d_in[3];
    const float* W1l  = (const float*)d_in[5];
    const float* b1l  = (const float*)d_in[6];
    const float* W1r  = (const float*)d_in[7];
    const float* W2l  = (const float*)d_in[8];
    const float* b2l  = (const float*)d_in[9];
    const float* W2r  = (const float*)d_in[10];
    const float* Wc1  = (const float*)d_in[11];
    const float* bc1  = (const float*)d_in[12];
    const float* Wc2  = (const float*)d_in[13];
    const float* bc2  = (const float*)d_in[14];

    const int N = in_sizes[0] / 128;
    const long long E = in_sizes[1] / 2;
    const int B = in_sizes[2];
    const int* src = eidx;
    const int* dst = eidx + E;

    // Workspace: agg[N*128] f32 | cnt[N] f32 | h[N*128] f32 | 4x bf16 weights
    float* agg = (float*)d_ws;
    float* cnt = agg + (size_t)N * 128;
    float* h   = cnt + N;
    __bf16* wb  = (__bf16*)(h + (size_t)N * 128);
    __bf16* w1l = wb;
    __bf16* w1r = wb + 16384;
    __bf16* w2l = wb + 32768;
    __bf16* w2r = wb + 49152;

    const int ZB = 2048;
    const long long eThreads = E * 32;
    const int eb = (int)((eThreads + 255) / 256);
    const int nTiles = (N + 15) / 16;
    const int gb = (nTiles + 7) / 8;

    // One-time (per call) weight conversion: 4 x 16384 elements, negligible.
    f32_to_bf16<<<64, 256, 0, stream>>>(W1l, w1l, 16384);
    f32_to_bf16<<<64, 256, 0, stream>>>(W1r, w1r, 16384);
    f32_to_bf16<<<64, 256, 0, stream>>>(W2l, w2l, 16384);
    f32_to_bf16<<<64, 256, 0, stream>>>(W2r, w2r, 16384);

    // ---- Layer 1 ----
    zero_f32<<<ZB, 256, 0, stream>>>(agg, (long long)N * 129);   // agg + cnt
    edge_scatter<<<eb, 256, 0, stream>>>(x, src, dst, agg, cnt, E, 1);
    sage_gemm<<<gb, 256, 0, stream>>>(agg, cnt, x, w1l, b1l, w1r, h, nTiles, N);

    // ---- Layer 2 (h2 aliases h1: each tile reads its rows before writing) ----
    zero_f32<<<ZB, 256, 0, stream>>>(agg, (long long)N * 128);
    edge_scatter<<<eb, 256, 0, stream>>>(h, src, dst, agg, cnt, E, 0);
    sage_gemm<<<gb, 256, 0, stream>>>(agg, cnt, h, w2l, b2l, w2r, h, nTiles, N);

    // ---- Head ----
    classifier<<<1, 128, 0, stream>>>(h, bat, tgt, Wc1, bc1, Wc2, bc2,
                                      (float*)d_out, N, B);
}
